// TextSim_71502615544408
// MI455X (gfx1250) — compile-verified, matrix operand order checked by reading, not run on hardware
//
#include <hip/hip_runtime.h>
#include <hip/hip_bf16.h>
#include <math.h>

// ---------------------------------------------------------------------------
// TextSim model on gfx1250 (MI455X).
// All dense math runs through v_wmma_f32_16x16x32_bf16 (fp32 accumulate).
// GEMM B-tiles staged into LDS by the Tensor Data Mover (tensor_load_to_lds,
// TENSORcnt-tracked). GRU recurrence: one WGP per direction, h double-
// buffered in LDS (ds_load/ds_store via integer ping-pong offsets); Whh
// streamed from L2 each step behind an opaque offset to block LICM.
// ---------------------------------------------------------------------------

typedef __attribute__((ext_vector_type(16))) __bf16 v16bf;
typedef __attribute__((ext_vector_type(8)))  float  v8f;
typedef __attribute__((ext_vector_type(4)))  unsigned int u32x4;
typedef __attribute__((ext_vector_type(8)))  int    i32x8;
typedef __attribute__((ext_vector_type(4)))  int    i32x4;

#define LSEQ 512
#define NB   64
#define BLROWS (NB * LSEQ)   // 32768

#if __has_builtin(__builtin_amdgcn_tensor_load_to_lds)
#define HAVE_TDM 1
#else
#define HAVE_TDM 0
#endif

static __device__ __forceinline__ float sigm(float x) { return 1.f / (1.f + __expf(-x)); }

// Opaque zero: defeats LICM/CSE for anything derived from it, costs one s_mov.
static __device__ __forceinline__ int opaque_zero() {
  int z;
  asm volatile("s_mov_b32 %0, 0" : "=s"(z));
  return z;
}

// Load a 16x32 bf16 WMMA operand fragment from a row-major [rows x ld] array.
// ISA 7.12.2 16-bit operand layout: lanes 0-15 hold K = k0+0..7 (v0..v3) and
// K = k0+16..23 (v4..v7) of row (lane&15); lanes 16-31 hold K = k0+8..15 and
// K = k0+24..31 of the same row.
static __device__ __forceinline__ v16bf load_frag16x32(const __bf16* base, int row0,
                                                       int ld, int k0, int lane) {
  int r  = row0 + (lane & 15);
  int kb = k0 + ((lane >> 4) << 3);
  const __bf16* p0 = base + (size_t)r * ld + kb;
  v16bf a;
#pragma unroll
  for (int i = 0; i < 8; ++i) { a[i] = p0[i]; a[i + 8] = p0[i + 16]; }
  return a;
}

#if HAVE_TDM
// Issue a TDM 2D tile load: rows x 32 bf16 tile (row stride ldb elements)
// from global into LDS at ldsdst, laid out [row][32] contiguously.
// D# layout per CDNA5 ISA (groups 0 and 1; 2D
// tensors need only 2 SGPR groups, groups 2/3 zero). This toolchain's builtin
// takes 6 args: (g0 x4, g1 x8, g2 x4, g3 x4, extra x8, cpol).
static __device__ __forceinline__ void tdm_load_tile_b(const __bf16* gsrc, void* ldsdst,
                                                       int rows, int ldb) {
  unsigned long long ga = (unsigned long long)(const void*)gsrc;
  unsigned lds = (unsigned)(size_t)ldsdst;        // generic LDS addr[31:0] = LDS byte addr
  u32x4 g0;
  g0[0] = 1u;                                     // count=1, user descriptor
  g0[1] = lds;                                    // lds_addr
  g0[2] = (unsigned)(ga & 0xFFFFFFFFu);           // global_addr[31:0]
  g0[3] = (unsigned)((ga >> 32) & 0x01FFFFFFu) | (2u << 30);  // addr[56:32] | type=2
  unsigned td0 = (unsigned)ldb;                   // tensor dim0 (elements)
  unsigned td1 = 0x40000000u;                     // dim1: effectively unbounded
  i32x8 g1;
  g1[0] = 0x00010000;                             // wg_mask=0, data_size=1 (2 bytes)
  g1[1] = (int)((td0 & 0xFFFFu) << 16);           // tensor_dim0[15:0]  -> bits 63:48
  g1[2] = (int)((td0 >> 16) | ((td1 & 0xFFFFu) << 16)); // dim0[31:16] | dim1[15:0]
  g1[3] = (int)((td1 >> 16) | (32u << 16));       // dim1[31:16] | tile_dim0=32
  g1[4] = (int)(unsigned)rows;                    // tile_dim1=rows, tile_dim2=0
  g1[5] = (int)td0;                               // tensor_dim0_stride[31:0] = ldb
  g1[6] = 0;                                      // stride0[47:32] | stride1[15:0]
  g1[7] = 0;                                      // stride1[47:16]
  i32x4 z4 = {};
  i32x8 z8 = {};
  __builtin_amdgcn_tensor_load_to_lds(g0, g1, z4, z4, z8, 0);
}
#endif

// ---------------------------------------------------------------------------
// Generic bf16 WMMA GEMM:  C[M,N] = A[M,K] * B^T (+bias) (+ReLU)
//   A: row-major M x K bf16, row stride lda, batch stride sA
//   B: if !BKMAJOR: weight-style (N x K) row-major, elem (k,n) at n*ldb + k
//      if  BKMAJOR: (K x N) row-major,              elem (k,n) at k*ldb + n
//   C: fp32, row stride ldc, batch stride sC
// Block: 128 threads (4 waves). Block tile 64(M) x 64(N); each wave owns a
// 16-row strip and 4 accumulator tiles; B tile staged into LDS by the TDM
// (weight-major path) or cooperative gather (K-major path).
// ---------------------------------------------------------------------------
template <bool BKMAJOR, bool RELU>
__global__ void k_gemm(const __bf16* __restrict__ A, const __bf16* __restrict__ Bw,
                       const float* __restrict__ bias, float* __restrict__ C,
                       int M, int N, int K, int lda, int ldb, int ldc,
                       long long sA, long long sB, long long sC) {
  __shared__ __bf16 bt[64 * 32];   // [n_local][k_local]
  int bz = blockIdx.z;
  A  += (size_t)bz * sA;
  Bw += (size_t)bz * sB;
  C  += (size_t)bz * sC;
  int tile_n = blockIdx.x * 64;
  int tile_m = blockIdx.y * 64;
  int tid  = threadIdx.x;
  int wave = tid >> 5, lane = tid & 31;
  int rowbase = tile_m + wave * 16;

  v8f acc[4] = {v8f{}, v8f{}, v8f{}, v8f{}};

  for (int k0 = 0; k0 < K; k0 += 32) {
    __syncthreads();   // protect bt from previous iteration's readers
    if (!BKMAJOR) {
#if HAVE_TDM
      // Tensor Data Mover: one DMA issue per block (only wave 0 reaches this;
      // waves with EXEC==0 branch around it), TENSORcnt-tracked.
      if (tid == 0) {
        tdm_load_tile_b(Bw + (size_t)tile_n * ldb + k0, bt, 64, ldb);
        __builtin_amdgcn_s_wait_tensorcnt(0);
      }
#else
      // contiguous along K: stage in 16-byte (8 x bf16) vector chunks
      for (int idx = tid; idx < 64 * 4; idx += 128) {
        int n = idx >> 2, kc = (idx & 3) * 8;
        u32x4 d = *(const u32x4*)(Bw + (size_t)(tile_n + n) * ldb + (k0 + kc));
        *(u32x4*)(bt + n * 32 + kc) = d;
      }
#endif
    } else {
      // K-major source: gather with 16-bit loads (transpose into [n][k])
      for (int idx = tid; idx < 64 * 32; idx += 128) {
        int n = idx >> 5, kk = idx & 31;
        bt[idx] = Bw[(size_t)(k0 + kk) * ldb + (tile_n + n)];
      }
    }
    __syncthreads();

    if (k0 + 32 < K)   // hint the next A strip into cache
      __builtin_prefetch(A + (size_t)(rowbase + (lane & 15)) * lda + k0 + 32, 0, 1);

    v16bf a = load_frag16x32(A, rowbase, lda, k0, lane);
#pragma unroll
    for (int nt = 0; nt < 4; ++nt) {
      v16bf b = load_frag16x32(bt, nt * 16, 32, 0, lane);
      acc[nt] = __builtin_amdgcn_wmma_f32_16x16x32_bf16(false, a, false, b,
                                                        (short)0, acc[nt], false, false);
    }
  }

  // C/D layout (ISA 7.12.2): VGPR i, lanes 0-15 -> (M=i, N=lane);
  // lanes 16-31 -> (M=i+8, N=lane-16).
  int n0 = lane & 15;
  int mofs = (lane >> 4) * 8;
#pragma unroll
  for (int nt = 0; nt < 4; ++nt) {
    int col = tile_n + nt * 16 + n0;
    float bv = bias ? bias[col] : 0.f;
#pragma unroll
    for (int i = 0; i < 8; ++i) {
      int row = rowbase + mofs + i;
      float v = acc[nt][i] + bv;
      if (RELU) v = fmaxf(v, 0.f);
      C[(size_t)row * ldc + col] = v;
    }
  }
}

// ---------------------------------------------------------------------------
// BiGRU scan, one workgroup per direction (gridDim.x == 2), 256 threads
// (8 waves). Batch (64) lives on the WMMA M dimension; hidden state h is
// double-buffered in LDS as fp32 (gate math) + bf16 (WMMA A operand). The
// ping-pong select is an *integer offset* from the single smem base so clang
// infers addrspace(3) and emits true ds_load/ds_store (not flat).
//   xg : (B*L, 6*HDIM) fp32, already contains x@Wih^T + bih for both dirs
//   Whh: (2, 3*HDIM, HDIM) bf16 ; bhh: (2, 3*HDIM) fp32
//   y  : (B, L, 2*HDIM) fp32 concat of both directions; ybf optional mirror
// ---------------------------------------------------------------------------
template <int HDIM>
__global__ void k_gru_scan(const float* __restrict__ xg, const __bf16* __restrict__ Whh,
                           const float* __restrict__ bhh, float* __restrict__ y,
                           __bf16* __restrict__ ybf, int L) {
  constexpr int NT    = HDIM / 16;  // gate-column tiles
  constexpr int TASKS = 4 * NT;     // 4 batch-row tiles x NT col tiles
  constexpr int TPW   = TASKS / 8;  // tasks per wave
  constexpr int HN    = 64 * HDIM;  // elements per h buffer
  extern __shared__ char smem[];
  float*  hfb = (float*)smem;                         // 2 x HN fp32
  __bf16* hbb = (__bf16*)(smem + (size_t)2 * HN * 4); // 2 x HN bf16

  int dir = blockIdx.x;
  const __bf16* W  = Whh + (size_t)dir * 3 * HDIM * HDIM;
  const float*  bhbase = bhh + dir * 3 * HDIM;
  int tid = threadIdx.x, wave = tid >> 5, lane = tid & 31;
  int n0 = lane & 15, mofs = (lane >> 4) * 8;
  const int XS = 6 * HDIM;

  for (int i = tid; i < 2 * HN; i += blockDim.x) {
    hfb[i] = 0.f; hbb[i] = (__bf16)0.f;
  }
  __syncthreads();

  for (int s = 0; s < L; ++s) {
    int ti = dir ? (L - 1 - s) : s;   // reversed traversal == reference's flip
    int po = (s & 1) * HN;            // old-buffer offset
    int pn = HN - po;                 // new-buffer offset

    // Opaque zero, re-derived each step: weight/bias addresses below depend on
    // a loop-varying SGPR the compiler cannot reason about, so their loads
    // stay inside the step loop (plain cached global loads, no spills).
    int zz = opaque_zero();
    const __bf16* Wt = W + zz;
    const float*  bh = bhbase + zz;

#pragma unroll
    for (int t = 0; t < TPW; ++t) {
      int task = wave * TPW + t;
      int mt = task / NT, jt = task % NT;
      v8f ar = v8f{}, az = v8f{}, an = v8f{};
#pragma unroll
      for (int kk = 0; kk < HDIM; kk += 32) {
        v16bf a  = load_frag16x32(hbb + po, mt * 16, HDIM, kk, lane);
        v16bf br = load_frag16x32(Wt, 0 * HDIM + jt * 16, HDIM, kk, lane);
        v16bf bz = load_frag16x32(Wt, 1 * HDIM + jt * 16, HDIM, kk, lane);
        v16bf bn = load_frag16x32(Wt, 2 * HDIM + jt * 16, HDIM, kk, lane);
        ar = __builtin_amdgcn_wmma_f32_16x16x32_bf16(false, a, false, br, (short)0, ar, false, false);
        az = __builtin_amdgcn_wmma_f32_16x16x32_bf16(false, a, false, bz, (short)0, az, false, false);
        an = __builtin_amdgcn_wmma_f32_16x16x32_bf16(false, a, false, bn, (short)0, an, false, false);
      }
      int j = jt * 16 + n0;
      float bhr = bh[j], bhz = bh[HDIM + j], bhn = bh[2 * HDIM + j];
#pragma unroll
      for (int i = 0; i < 8; ++i) {
        int b = mt * 16 + mofs + i;
        const float* xrow = xg + ((size_t)b * L + ti) * XS + dir * 3 * HDIM;
        float r = sigm(xrow[j] + ar[i] + bhr);
        float z = sigm(xrow[HDIM + j] + az[i] + bhz);
        float n = tanhf(xrow[2 * HDIM + j] + r * (an[i] + bhn));
        float ho = hfb[po + b * HDIM + j];
        float v = (1.f - z) * n + z * ho;
        hfb[pn + b * HDIM + j] = v;
        hbb[pn + b * HDIM + j] = (__bf16)v;
        size_t o = ((size_t)b * L + ti) * (2 * HDIM) + dir * HDIM + j;
        y[o] = v;
        if (ybf) ybf[o] = (__bf16)v;
      }
    }
    __syncthreads();   // publish the new buffer before next step reads it
  }
}

// ---------------------------------------------------------------------------
// Small helper kernels
// ---------------------------------------------------------------------------
__global__ void k_cvt_bf16(const float* __restrict__ src, __bf16* __restrict__ dst, long long n) {
  long long i = (long long)blockIdx.x * blockDim.x + threadIdx.x;
  if (i < n) dst[i] = (__bf16)src[i];
}

__global__ void k_embed(const int* __restrict__ idx, const float* __restrict__ emb,
                        float* __restrict__ e, __bf16* __restrict__ eb) {
  int row = blockIdx.x, c = threadIdx.x;                  // 128 threads
  float v = emb[(size_t)idx[row] * 128 + c];
  e[(size_t)row * 128 + c]  = v;
  eb[(size_t)row * 128 + c] = (__bf16)v;
}

__global__ void k_rowdot(const float* __restrict__ x, const float* __restrict__ w,
                         float* __restrict__ out, int D) {
  __shared__ float red[256];
  int row = blockIdx.x, tid = threadIdx.x;
  float p = 0.f;
  for (int d = tid; d < D; d += blockDim.x) p += x[(size_t)row * D + d] * w[d];
  red[tid] = p; __syncthreads();
  for (int st = 128; st > 0; st >>= 1) { if (tid < st) red[tid] += red[tid + st]; __syncthreads(); }
  if (tid == 0) out[row] = red[0];
}

__global__ void k_qm(const float* __restrict__ q, const float* __restrict__ wm,
                     __bf16* __restrict__ qmb, long long n) {
  long long i = (long long)blockIdx.x * blockDim.x + threadIdx.x;
  if (i < n) qmb[i] = (__bf16)(q[i] * wm[i & 255]);
}

// attn row softmax with additive terms + key mask. Row = b*LQ + i, 128 thr.
__global__ void k_softmax(float* __restrict__ attn, const float* __restrict__ aq,
                          const float* __restrict__ ak, const int* __restrict__ in2,
                          __bf16* __restrict__ pbf) {
  __shared__ float red[128];
  int row = blockIdx.x, b = row >> 9, tid = threadIdx.x;
  float* arow = attn + (size_t)row * 512;
  float vq = aq[row];
  float vals[4];
  float mx = -INFINITY;
#pragma unroll
  for (int t = 0; t < 4; ++t) {
    int j = tid + t * 128;
    float v = (in2[b * 512 + j] == 0) ? -INFINITY : (arow[j] + vq + ak[b * 512 + j]);
    vals[t] = v; mx = fmaxf(mx, v);
  }
  red[tid] = mx; __syncthreads();
  for (int st = 64; st > 0; st >>= 1) { if (tid < st) red[tid] = fmaxf(red[tid], red[tid + st]); __syncthreads(); }
  mx = red[0]; __syncthreads();
  float sm = 0.f;
#pragma unroll
  for (int t = 0; t < 4; ++t) {
    float e = (vals[t] == -INFINITY) ? 0.f : __expf(vals[t] - mx);
    vals[t] = e; sm += e;
  }
  red[tid] = sm; __syncthreads();
  for (int st = 64; st > 0; st >>= 1) { if (tid < st) red[tid] += red[tid + st]; __syncthreads(); }
  float inv = (red[0] > 0.f) ? 1.f / red[0] : 0.f;
#pragma unroll
  for (int t = 0; t < 4; ++t) {
    int j = tid + t * 128;
    float p = vals[t] * inv;
    arow[j] = p;
    pbf[(size_t)row * 512 + j] = (__bf16)p;
  }
}

__global__ void k_concat_s(const float* __restrict__ q, const float* __restrict__ v,
                           float* __restrict__ s, __bf16* __restrict__ sb, long long n) {
  long long i = (long long)blockIdx.x * blockDim.x + threadIdx.x;
  if (i >= n) return;
  long long row = i >> 9; int c = (int)(i & 511);
  float val = (c < 256) ? q[row * 256 + c] : v[row * 256 + (c - 256)];
  s[i] = val; sb[i] = (__bf16)val;
}

__global__ void k_pool(const float* __restrict__ s, const float* __restrict__ m,
                       float* __restrict__ pooled, __bf16* __restrict__ pb) {
  int b = blockIdx.x, tid = threadIdx.x;          // 256 threads
  for (int c = tid; c < 1024; c += 256) {
    const float* src = (c < 512) ? (s + (size_t)b * 512 * 512 + c)
                                 : (m + (size_t)b * 512 * 512 + (c - 512));
    float mx = -INFINITY;
    for (int l = 0; l < 512; ++l) mx = fmaxf(mx, src[(size_t)l * 512]);
    pooled[b * 1024 + c] = mx;
    pb[b * 1024 + c]     = (__bf16)mx;
  }
}

__global__ void k_out(const float* __restrict__ h, const float* __restrict__ W,
                      const float* __restrict__ bo, float* __restrict__ out) {
  __shared__ float red[2][256];
  int b = blockIdx.x, tid = threadIdx.x;          // 256 threads
  float p0 = 0.f, p1 = 0.f;
  for (int c = tid; c < 1024; c += 256) {
    float hv = h[b * 1024 + c];
    p0 += hv * W[c];
    p1 += hv * W[1024 + c];
  }
  red[0][tid] = p0; red[1][tid] = p1; __syncthreads();
  for (int st = 128; st > 0; st >>= 1) {
    if (tid < st) { red[0][tid] += red[0][tid + st]; red[1][tid] += red[1][tid + st]; }
    __syncthreads();
  }
  if (tid < 2) out[b * 2 + tid] = red[tid][0] + bo[tid];
}

// ---------------------------------------------------------------------------
extern "C" void kernel_launch(void* const* d_in, const int* in_sizes, int n_in,
                              void* d_out, int out_size, void* d_ws, size_t ws_size,
                              hipStream_t stream) {
  const int*   input1  = (const int*)d_in[0];
  const int*   input2  = (const int*)d_in[1];
  const float* embed   = (const float*)d_in[2];
  const float* eWih0   = (const float*)d_in[3];
  const float* eWhh0   = (const float*)d_in[4];
  const float* ebih0   = (const float*)d_in[5];
  const float* ebhh0   = (const float*)d_in[6];
  const float* eWih1   = (const float*)d_in[7];
  const float* eWhh1   = (const float*)d_in[8];
  const float* ebih1   = (const float*)d_in[9];
  const float* ebhh1   = (const float*)d_in[10];
  const float* fWih0   = (const float*)d_in[11];
  const float* fWhh0   = (const float*)d_in[12];
  const float* fbih0   = (const float*)d_in[13];
  const float* fbhh0   = (const float*)d_in[14];
  const float* fWih1   = (const float*)d_in[15];
  const float* fWhh1   = (const float*)d_in[16];
  const float* fbih1   = (const float*)d_in[17];
  const float* fbhh1   = (const float*)d_in[18];
  const float* attnw   = (const float*)d_in[19];
  const float* mlpW    = (const float*)d_in[20];
  const float* mlpb    = (const float*)d_in[21];
  const float* outW    = (const float*)d_in[22];
  const float* outb    = (const float*)d_in[23];
  float* out = (float*)d_out;

  // ---- workspace carve-out ----
  char* wsb = (char*)d_ws;
  size_t off = 0;
  auto carve = [&](size_t bytes) -> void* {
    void* p = wsb + off;
    off += (bytes + 255) & ~(size_t)255;
    return p;
  };
  __bf16* wih0b  = (__bf16*)carve((size_t)768 * 128 * 2);
  __bf16* whh0b  = (__bf16*)carve((size_t)768 * 128 * 2);
  __bf16* wih1b  = (__bf16*)carve((size_t)768 * 256 * 2);
  __bf16* whh1b  = (__bf16*)carve((size_t)768 * 128 * 2);
  __bf16* fwih0b = (__bf16*)carve((size_t)1536 * 512 * 2);
  __bf16* fwhh0b = (__bf16*)carve((size_t)1536 * 256 * 2);
  __bf16* fwih1b = (__bf16*)carve((size_t)1536 * 512 * 2);
  __bf16* fwhh1b = (__bf16*)carve((size_t)1536 * 256 * 2);
  __bf16* mlpwb  = (__bf16*)carve((size_t)1024 * 1024 * 2);
  float*  ef     = (float*) carve((size_t)BLROWS * 128 * 4);
  __bf16* eb     = (__bf16*)carve((size_t)BLROWS * 128 * 2);
  float*  xg     = (float*) carve((size_t)BLROWS * 1536 * 4);   // shared GEMM scratch
  float*  y0f    = (float*) carve((size_t)BLROWS * 256 * 4);
  __bf16* y0b    = (__bf16*)carve((size_t)BLROWS * 256 * 2);
  float*  qf     = (float*) carve((size_t)BLROWS * 256 * 4);
  float*  kf     = (float*) carve((size_t)BLROWS * 256 * 4);
  __bf16* kbb    = (__bf16*)carve((size_t)BLROWS * 256 * 2);
  float*  aq     = (float*) carve((size_t)BLROWS * 4);
  float*  ak     = (float*) carve((size_t)BLROWS * 4);
  __bf16* qmb    = (__bf16*)carve((size_t)BLROWS * 256 * 2);
  float*  attnf  = (float*) carve((size_t)NB * 512 * 512 * 4);
  __bf16* attnb  = (__bf16*)carve((size_t)NB * 512 * 512 * 2);
  float*  vf     = (float*) carve((size_t)BLROWS * 256 * 4);
  float*  sf     = (float*) carve((size_t)BLROWS * 512 * 4);
  __bf16* sb     = (__bf16*)carve((size_t)BLROWS * 512 * 2);
  float*  m0f    = (float*) carve((size_t)BLROWS * 512 * 4);
  __bf16* m0b    = (__bf16*)carve((size_t)BLROWS * 512 * 2);
  float*  mf     = (float*) carve((size_t)BLROWS * 512 * 4);
  float*  poolf  = (float*) carve((size_t)NB * 1024 * 4);
  __bf16* poolb  = (__bf16*)carve((size_t)NB * 1024 * 2);
  float*  hmlp   = (float*) carve((size_t)NB * 1024 * 4);

  auto cvt = [&](const float* s, __bf16* d, long long n) {
    k_cvt_bf16<<<(unsigned)((n + 255) / 256), 256, 0, stream>>>(s, d, n);
  };
  cvt(eWih0, wih0b, 768LL * 128);
  cvt(eWhh0, whh0b, 768LL * 128);
  cvt(eWih1, wih1b, 768LL * 256);
  cvt(eWhh1, whh1b, 768LL * 128);
  cvt(fWih0, fwih0b, 1536LL * 512);
  cvt(fWhh0, fwhh0b, 1536LL * 256);
  cvt(fWih1, fwih1b, 1536LL * 512);
  cvt(fWhh1, fwhh1b, 1536LL * 256);
  cvt(mlpW,  mlpwb, 1024LL * 1024);

  // double-buffered h: 12 bytes per element total (2x fp32 + 2x bf16)
  const int shm128 = 64 * 128 * 12;   //  96 KB
  const int shm256 = 64 * 256 * 12;   // 192 KB (of 320 KB WGP LDS)
  (void)hipFuncSetAttribute((const void*)k_gru_scan<128>,
                            hipFuncAttributeMaxDynamicSharedMemorySize, shm128);
  (void)hipFuncSetAttribute((const void*)k_gru_scan<256>,
                            hipFuncAttributeMaxDynamicSharedMemorySize, shm256);

  // ---- encoder path, run for each of the two sequences ----
  auto encode = [&](const int* idx, float* outy, __bf16* outyb) {
    k_embed<<<BLROWS, 128, 0, stream>>>(idx, embed, ef, eb);
    // layer0: xg = e @ Wih0^T + bih0   (M=32768, N=768, K=128)
    k_gemm<false, false><<<dim3(12, 512, 1), 128, 0, stream>>>(
        eb, wih0b, ebih0, xg, BLROWS, 768, 128, 128, 128, 768, 0, 0, 0);
    k_gru_scan<128><<<2, 256, shm128, stream>>>(xg, whh0b, ebhh0, y0f, y0b, LSEQ);
    // layer1: xg = y0 @ Wih1^T + bih1  (K=256)
    k_gemm<false, false><<<dim3(12, 512, 1), 128, 0, stream>>>(
        y0b, wih1b, ebih1, xg, BLROWS, 768, 256, 256, 256, 768, 0, 0, 0);
    k_gru_scan<128><<<2, 256, shm128, stream>>>(xg, whh1b, ebhh1, outy, outyb, LSEQ);
  };
  encode(input1, qf, nullptr);
  encode(input2, kf, kbb);

  // ---- attention ----
  k_rowdot<<<BLROWS, 256, 0, stream>>>(qf, attnw, aq, 256);
  k_rowdot<<<BLROWS, 256, 0, stream>>>(kf, attnw + 256, ak, 256);
  k_qm<<<(unsigned)(((long long)BLROWS * 256 + 255) / 256), 256, 0, stream>>>(
      qf, attnw + 512, qmb, (long long)BLROWS * 256);
  // bilinear: attn[b] = (q*wm)[b] @ k[b]^T  (batched M=512 N=512 K=256)
  k_gemm<false, false><<<dim3(8, 8, NB), 128, 0, stream>>>(
      qmb, kbb, nullptr, attnf, 512, 512, 256, 256, 256, 512,
      512LL * 256, 512LL * 256, 512LL * 512);
  k_softmax<<<BLROWS, 128, 0, stream>>>(attnf, aq, ak, input2, attnb);
  // v[b] = probs[b] @ k[b]   (B is K-major here; M=512 N=256 K=512)
  k_gemm<true, false><<<dim3(4, 8, NB), 128, 0, stream>>>(
      attnb, kbb, nullptr, vf, 512, 256, 512, 512, 256, 256,
      512LL * 512, 512LL * 256, 512LL * 256);
  k_concat_s<<<(unsigned)(((long long)BLROWS * 512 + 255) / 256), 256, 0, stream>>>(
      qf, vf, sf, sb, (long long)BLROWS * 512);

  // ---- fusion BiGRU stack (H2 = 256) ----
  k_gemm<false, false><<<dim3(24, 512, 1), 128, 0, stream>>>(
      sb, fwih0b, fbih0, xg, BLROWS, 1536, 512, 512, 512, 1536, 0, 0, 0);
  k_gru_scan<256><<<2, 256, shm256, stream>>>(xg, fwhh0b, fbhh0, m0f, m0b, LSEQ);
  k_gemm<false, false><<<dim3(24, 512, 1), 128, 0, stream>>>(
      m0b, fwih1b, fbih1, xg, BLROWS, 1536, 512, 512, 512, 1536, 0, 0, 0);
  k_gru_scan<256><<<2, 256, shm256, stream>>>(xg, fwhh1b, fbhh1, mf, nullptr, LSEQ);

  // ---- pool + MLP + head ----
  k_pool<<<NB, 256, 0, stream>>>(sf, mf, poolf, poolb);
  k_gemm<false, true><<<dim3(16, 1, 1), 128, 0, stream>>>(
      poolb, mlpwb, mlpb, hmlp, 64, 1024, 1024, 1024, 1024, 1024, 0, 0, 0);
  k_out<<<NB, 256, 0, stream>>>(hmlp, outW, outb, out);

  (void)in_sizes; (void)n_in; (void)out_size; (void)ws_size;
}
